// CustomCosineSimCodebook_19396072309113
// MI455X (gfx1250) — matrix operand order repaired and checked by reading, not compiled, not run on hardware
//
#include <hip/hip_runtime.h>
#include <hip/hip_bf16.h>

// Problem dims (fixed by reference): h=1, b=8, n=2048, d=512, c=8192
#define NROWS 16384   // b*n
#define DIM   512     // d
#define NCOLS 8192    // c

typedef __attribute__((ext_vector_type(16))) __bf16 v16bf;
typedef __attribute__((ext_vector_type(8)))  __bf16 v8bf;
typedef __attribute__((ext_vector_type(8)))  float  v8f;
typedef __attribute__((ext_vector_type(4)))  float  v4f;

union FragU { v16bf v; v8bf h[2]; };

__device__ __forceinline__ void split_bf16(float f, __bf16* hi, __bf16* lo) {
    // round-to-nearest-even f32 -> bf16, then bf16 of the residual
    unsigned int u  = __builtin_bit_cast(unsigned int, f);
    unsigned int hb = (u + 0x7FFFu + ((u >> 16) & 1u)) >> 16;
    float hf        = __builtin_bit_cast(float, hb << 16);
    float lf        = f - hf;
    unsigned int ul = __builtin_bit_cast(unsigned int, lf);
    unsigned int lb = (ul + 0x7FFFu + ((ul >> 16) & 1u)) >> 16;
    *hi = __builtin_bit_cast(__bf16, (unsigned short)hb);
    *lo = __builtin_bit_cast(__bf16, (unsigned short)lb);
}

// A fragment, 16-bit 16x32 layout: lane half lh reads k = 8*lh..8*lh+7 and 16+8*lh..+7
__device__ __forceinline__ v16bf load_fragA(const __bf16* p) {
    FragU u;
    u.h[0] = *(const v8bf*)(p);
    u.h[1] = *(const v8bf*)(p + 16);
    return u.v;
}
// B fragment, 32x16 layout: lane half lh reads k = 8*lh..8*lh+7 and 16+8*lh..+7
__device__ __forceinline__ v16bf load_fragB(const __bf16* p) {
    FragU u;
    u.h[0] = *(const v8bf*)(p);
    u.h[1] = *(const v8bf*)(p + 16);
    return u.v;
}

__global__ void vq_init_keys(unsigned long long* keys) {
    int i = blockIdx.x * blockDim.x + threadIdx.x;
    if (i < NROWS) keys[i] = 0ull;
}

__global__ __launch_bounds__(256)
void vq_gemm_argmax(const float* __restrict__ X, const float* __restrict__ E,
                    float* __restrict__ dist, unsigned long long* __restrict__ keys)
{
    __shared__ __bf16 Ah[128][32];
    __shared__ __bf16 Al[128][32];
    __shared__ __bf16 Bh[128][32];
    __shared__ __bf16 Bl[128][32];

    const int t     = threadIdx.x;
    const int cBase = blockIdx.x * 128;
    const int mBase = blockIdx.y * 128;
    const int wave  = t >> 5;
    const int lane  = t & 31;
    const int wm    = wave & 3;   // 4 waves over M (32 rows each)
    const int wn    = wave >> 2;  // 2 waves over N (64 cols each)
    const int lh    = lane >> 4;  // 16-lane half
    const int l16   = lane & 15;

    // per-thread staging assignment (4 float4 of A tile, 4 of B tile per chunk)
    int srow[4], skq[4];
#pragma unroll
    for (int i = 0; i < 4; ++i) {
        int idx = t + i * 256;          // 0..1023
        srow[i] = idx >> 3;             // 0..127
        skq[i]  = (idx & 7) << 2;       // 0,4,..,28
    }

    v8f acc[2][4];
#pragma unroll
    for (int i = 0; i < 2; ++i)
#pragma unroll
        for (int j = 0; j < 4; ++j) {
            v8f z = {0.f, 0.f, 0.f, 0.f, 0.f, 0.f, 0.f, 0.f};
            acc[i][j] = z;
        }

    // ---- software pipeline: prefetch chunk 0 into registers ----
    v4f pa[4], pb[4];
#pragma unroll
    for (int i = 0; i < 4; ++i) {
        pa[i] = *(const v4f*)(X + (size_t)(mBase + srow[i]) * DIM + skq[i]);
        pb[i] = *(const v4f*)(E + (size_t)(cBase + srow[i]) * DIM + skq[i]);
    }

    for (int k0 = 0; k0 < DIM; k0 += 32) {
        // convert previously-prefetched chunk and store to LDS
#pragma unroll
        for (int i = 0; i < 4; ++i) {
#pragma unroll
            for (int j = 0; j < 4; ++j) {
                split_bf16(pa[i][j], &Ah[srow[i]][skq[i] + j], &Al[srow[i]][skq[i] + j]);
                split_bf16(pb[i][j], &Bh[srow[i]][skq[i] + j], &Bl[srow[i]][skq[i] + j]);
            }
        }
        __syncthreads();

        // prefetch next chunk: global loads overlap the WMMA phase below
        if (k0 + 32 < DIM) {
#pragma unroll
            for (int i = 0; i < 4; ++i) {
                pa[i] = *(const v4f*)(X + (size_t)(mBase + srow[i]) * DIM + k0 + 32 + skq[i]);
                pb[i] = *(const v4f*)(E + (size_t)(cBase + srow[i]) * DIM + k0 + 32 + skq[i]);
            }
        }

        // Load A fragments for this wave's two 16-row tiles.
        v16bf a_h[2], a_l[2];
#pragma unroll
        for (int mt = 0; mt < 2; ++mt) {
            int r = wm * 32 + mt * 16 + l16;
            a_h[mt] = load_fragA(&Ah[r][8 * lh]);
            a_l[mt] = load_fragA(&Al[r][8 * lh]);
        }

#pragma unroll
        for (int nt = 0; nt < 4; ++nt) {
            int c = wn * 64 + nt * 16 + l16;
            v16bf b_h = load_fragB(&Bh[c][8 * lh]);
            v16bf b_l = load_fragB(&Bl[c][8 * lh]);
#pragma unroll
            for (int mt = 0; mt < 2; ++mt) {
                // dot ~= lo*hi + hi*lo + hi*hi  (fp32-class accuracy on bf16 XDL path)
                acc[mt][nt] = __builtin_amdgcn_wmma_f32_16x16x32_bf16(
                    false, a_l[mt], false, b_h, (short)0, acc[mt][nt], false, false);
                acc[mt][nt] = __builtin_amdgcn_wmma_f32_16x16x32_bf16(
                    false, a_h[mt], false, b_l, (short)0, acc[mt][nt], false, false);
                acc[mt][nt] = __builtin_amdgcn_wmma_f32_16x16x32_bf16(
                    false, a_h[mt], false, b_h, (short)0, acc[mt][nt], false, false);
            }
        }
        __syncthreads();   // protect LDS before next iteration's stores
    }

    // Epilogue 1: stream dist out (non-temporal: 537 MB >> 192 MB L2).
    // C/D layout: value acc[mt][nt][v] is at (M = mt*16 + v + 8*lh, N = nt*16 + l16).
#pragma unroll
    for (int mt = 0; mt < 2; ++mt)
#pragma unroll
        for (int v = 0; v < 8; ++v) {
            int row = mBase + wm * 32 + mt * 16 + v + 8 * lh;
#pragma unroll
            for (int nt = 0; nt < 4; ++nt) {
                int col = cBase + wn * 64 + nt * 16 + l16;
                __builtin_nontemporal_store(acc[mt][nt][v],
                                            dist + (size_t)row * NCOLS + col);
            }
        }

    // Epilogue 2: fused per-row argmax over this wave's 64 columns -> global atomicMax
    // key = (monotonic(f32) << 32) | (0xFFFFFFFF - col)  => max key == max val, ties -> min col.
#pragma unroll
    for (int mt = 0; mt < 2; ++mt) {
#pragma unroll
        for (int v = 0; v < 8; ++v) {
            float best = acc[mt][0][v];
            int   bcol = cBase + wn * 64 + l16;
#pragma unroll
            for (int nt = 1; nt < 4; ++nt) {
                float val = acc[mt][nt][v];
                int   col = cBase + wn * 64 + nt * 16 + l16;
                if (val > best || (val == best && col < bcol)) { best = val; bcol = col; }
            }
            // reduce across the 16 lanes of this half (same row, different N)
#pragma unroll
            for (int m = 8; m >= 1; m >>= 1) {
                float oval = __shfl_xor(best, m, 32);
                int   ocol = __shfl_xor(bcol, m, 32);
                if (oval > best || (oval == best && ocol < bcol)) { best = oval; bcol = ocol; }
            }
            if (l16 == v) {
                unsigned int ub = __builtin_bit_cast(unsigned int, best);
                ub ^= (unsigned int)(((int)ub) >> 31) | 0x80000000u;
                unsigned long long key =
                    ((unsigned long long)ub << 32) |
                    (unsigned long long)(0xFFFFFFFFu - (unsigned int)bcol);
                int row = mBase + wm * 32 + mt * 16 + v + 8 * lh;
                atomicMax(&keys[row], key);
            }
        }
    }
}

__global__ __launch_bounds__(128)
void vq_finish(const unsigned long long* __restrict__ keys,
               const float* __restrict__ E,
               float* __restrict__ quant, float* __restrict__ indf)
{
    int row = blockIdx.x;
    unsigned long long key = keys[row];
    unsigned int col = 0xFFFFFFFFu - (unsigned int)(key & 0xFFFFFFFFull);
    if (threadIdx.x == 0) indf[row] = (float)col;
    const v4f* s4 = (const v4f*)(E + (size_t)col * DIM);
    v4f*       d4 = (v4f*)(quant + (size_t)row * DIM);
    d4[threadIdx.x] = s4[threadIdx.x];   // 128 threads x 16B = 512 floats
}

extern "C" void kernel_launch(void* const* d_in, const int* in_sizes, int n_in,
                              void* d_out, int out_size, void* d_ws, size_t ws_size,
                              hipStream_t stream) {
    const float* X = (const float*)d_in[0];   // (8,2048,512) f32
    const float* E = (const float*)d_in[1];   // (1,8192,512) f32 (h=1)
    float* out   = (float*)d_out;
    float* quant = out;                                   // 8*2048*512
    float* indf  = out + (size_t)NROWS * DIM;             // 16384
    float* dist  = indf + NROWS;                          // 16384*8192
    unsigned long long* keys = (unsigned long long*)d_ws; // 128 KB

    vq_init_keys<<<(NROWS + 255) / 256, 256, 0, stream>>>(keys);
    dim3 grid(NCOLS / 128, NROWS / 128);  // (64, 128)
    vq_gemm_argmax<<<grid, 256, 0, stream>>>(X, E, dist, keys);
    vq_finish<<<NROWS, 128, 0, stream>>>(keys, E, quant, indf);
}
